// MultiHeadAttention_17935783428114
// MI455X (gfx1250) — matrix operand order repaired and checked by reading, not compiled, not run on hardware
//
#include <hip/hip_runtime.h>

typedef __attribute__((ext_vector_type(16))) __bf16 v16bf;
typedef __attribute__((ext_vector_type(8)))  __bf16 v8bf;
typedef __attribute__((ext_vector_type(8)))  float  v8f;

#define D_MODEL 1024
#define SEQ     4096
#define NHEAD   16
#define DHEAD   64
#define BATCH   2
#define ROWS    (BATCH * SEQ)   // 8192

// A-fragment (16x32 bf16) element->K mapping per CDNA5 ISA table.
__device__ __forceinline__ int k_of(int e, int hi) {
    return (e & 7) + ((e & 8) << 1) + (hi << 3);
}

__device__ __forceinline__ v8f wmma_bf16(v16bf a, v16bf b, v8f c) {
    return __builtin_amdgcn_wmma_f32_16x16x32_bf16(false, a, false, b, (short)0, c, false, false);
}

// CDNA5 async global->LDS 128-bit copy (ASYNCcnt, no VGPR round-trip).
__device__ __forceinline__ void async_ld16_off(unsigned ldsoff, const void* g) {
    asm volatile("global_load_async_to_lds_b128 %0, %1, off"
                 :: "v"(ldsoff), "v"(g) : "memory");
}
#define WAIT_ASYNC(n) asm volatile("s_wait_asynccnt " #n ::: "memory")

__device__ __forceinline__ unsigned lds_off(const void* p) {
    return (unsigned)(unsigned long long)p;   // low 32 bits of generic LDS ptr = offset
}

// ---------------- Phase 0: convert + transpose prep ----------------

__global__ void cvt_bf16_kernel(const float* __restrict__ x, __bf16* __restrict__ y) {
    int i = blockIdx.x * blockDim.x + threadIdx.x;
    float4 a = ((const float4*)x)[2 * i];
    float4 b = ((const float4*)x)[2 * i + 1];
    v8bf o;
    o[0] = (__bf16)a.x; o[1] = (__bf16)a.y; o[2] = (__bf16)a.z; o[3] = (__bf16)a.w;
    o[4] = (__bf16)b.x; o[5] = (__bf16)b.y; o[6] = (__bf16)b.z; o[7] = (__bf16)b.w;
    ((v8bf*)y)[i] = o;
}

__global__ void transpose_bf16_kernel(const float* __restrict__ w, __bf16* __restrict__ wt) {
    __shared__ float tile[32][33];
    int m0 = blockIdx.x * 32, d0 = blockIdx.y * 32;
    tile[threadIdx.y][threadIdx.x] = w[(size_t)(m0 + threadIdx.y) * D_MODEL + d0 + threadIdx.x];
    __syncthreads();
    wt[(size_t)(d0 + threadIdx.y) * D_MODEL + m0 + threadIdx.x] = (__bf16)tile[threadIdx.x][threadIdx.y];
}

// ---------------- Phase 1: fused QKV projection + RoPE ----------------

__global__ void __launch_bounds__(384) qkv_rope_kernel(
    const __bf16* __restrict__ Xb,
    const __bf16* __restrict__ WqT, const __bf16* __restrict__ WkT, const __bf16* __restrict__ WvT,
    const int* __restrict__ pos_ids,
    __bf16* __restrict__ Qb,   // [B,H,S,DH]
    __bf16* __restrict__ Kt,   // [B,H,DH,S]
    __bf16* __restrict__ Vb)   // [B,H,S,DH]
{
    __shared__ alignas(16) __bf16 Xs[2][64][32];
    __shared__ alignas(16) __bf16 Ws[2][3][32][64];
    __shared__ alignas(16) __bf16 Ktile[4][64][16];

    const int row0 = blockIdx.x * 64;
    const int h    = blockIdx.y;
    const int tid  = threadIdx.x;
    const int wave = tid >> 5, lane = tid & 31;
    const int mat  = wave >> 2;
    const int ssub = wave & 3;
    const int hi   = lane >> 4, ln = lane & 15;

    // Precompute per-thread staging slots (3 async b128 per thread per k-step).
    const char* gsrc[3];
    unsigned    gstep[3];
    unsigned    lbase[3];
    unsigned    lstride[3];
    #pragma unroll
    for (int ii = 0; ii < 3; ++ii) {
        int idx = tid + ii * 384;
        int j = (idx >= 1024) ? (idx - 1024) : idx;   // duplicate 128 slots, benign
        if (j < 256) {                                // X tile 64x32
            int r = j >> 2, c = j & 3;
            gsrc[ii]    = (const char*)&Xb[(size_t)(row0 + r) * D_MODEL + c * 8];
            gstep[ii]   = 32 * 2;                     // advance 32 bf16 along K
            lbase[ii]   = lds_off(&((uint4*)&Xs[0][0][0])[j]);
            lstride[ii] = sizeof(Xs[0]);              // 4096
        } else {                                      // W tiles 3x32x64
            int w = j - 256;
            int mm = w >> 8, rem = w & 255;
            int r = rem >> 3, c = rem & 7;
            const __bf16* src = (mm == 0) ? WqT : ((mm == 1) ? WkT : WvT);
            gsrc[ii]    = (const char*)&src[(size_t)r * D_MODEL + h * DHEAD + c * 8];
            gstep[ii]   = 32 * D_MODEL * 2;           // advance 32 rows
            lbase[ii]   = lds_off(&((uint4*)&Ws[0][0][0][0])[w]);
            lstride[ii] = sizeof(Ws[0]);              // 12288
        }
    }
    auto issue = [&](int buf) {
        #pragma unroll
        for (int ii = 0; ii < 3; ++ii) {
            async_ld16_off(lbase[ii] + (buf ? lstride[ii] : 0), gsrc[ii]);
            gsrc[ii] += gstep[ii];
        }
    };

    v8f acc[4];
    #pragma unroll
    for (int t = 0; t < 4; ++t)
        #pragma unroll
        for (int r = 0; r < 8; ++r) acc[t][r] = 0.0f;

    issue(0);
    for (int kt = 0; kt < D_MODEL / 32; ++kt) {
        const int cur = kt & 1;
        if (kt < D_MODEL / 32 - 1) {
            issue(cur ^ 1);
            WAIT_ASYNC(0x3);
        } else {
            WAIT_ASYNC(0x0);
        }
        __syncthreads();

        v16bf a;
        #pragma unroll
        for (int e = 0; e < 16; ++e) a[e] = Xs[cur][ssub * 16 + ln][k_of(e, hi)];
        #pragma unroll
        for (int t = 0; t < 4; ++t) {
            v16bf b;
            #pragma unroll
            for (int e = 0; e < 16; ++e) b[e] = Ws[cur][mat][k_of(e, hi)][t * 16 + ln];
            acc[t] = wmma_bf16(a, b, acc[t]);
        }
        __syncthreads();
    }

    // Epilogue: RoPE (Q,K) + stores
    const int grow0 = row0 + ssub * 16;
    const int b_ = grow0 >> 12;
    const int s0 = grow0 & 4095;
    const size_t bh = (size_t)(b_ * NHEAD + h);

    #pragma unroll
    for (int t = 0; t < 4; ++t) {
        #pragma unroll
        for (int r = 0; r < 8; ++r) {
            int s_ = s0 + r + hi * 8;
            int d  = t * 16 + ln;
            float val = acc[t][r];
            if (mat < 2) {
                int pos = pos_ids[s_];
                int i = d >> 1;
                float freq = __expf(-0.28782313662f * (float)i);   // theta^(-i/32)
                float ang = (float)pos * freq;
                float sn, cs;
                __sincosf(ang, &sn, &cs);
                float other = __shfl_xor(val, 1);
                val = (d & 1) ? (other * sn + val * cs)
                              : (val * cs - other * sn);
            }
            if (mat == 0)      Qb[(bh * SEQ + s_) * DHEAD + d] = (__bf16)val;
            else if (mat == 1) Ktile[ssub][d][r + hi * 8] = (__bf16)val;
            else               Vb[(bh * SEQ + s_) * DHEAD + d] = (__bf16)val;
        }
    }
    if (mat == 1) {
        #pragma unroll
        for (int rr = 0; rr < 2; ++rr) {
            int d = lane + rr * 32;
            const uint4* srow = (const uint4*)&Ktile[ssub][d][0];
            uint4* dst = (uint4*)&Kt[(bh * DHEAD + d) * SEQ + s0];
            dst[0] = srow[0];
            dst[1] = srow[1];
        }
    }
}

// ---------------- Phase 2: flash attention (causal, online softmax) ----------------

__global__ void __launch_bounds__(128) attn_kernel(
    const __bf16* __restrict__ Qb, const __bf16* __restrict__ Kt,
    const __bf16* __restrict__ Vb, __bf16* __restrict__ Ob)
{
    __shared__ alignas(16) __bf16 Kls[2][64][32];
    __shared__ alignas(16) __bf16 Vls[2][32][64];
    __shared__ alignas(16) __bf16 Pls[4][16][32];

    const int q0blk = blockIdx.x * 64;
    const int h = blockIdx.y, b_ = blockIdx.z;
    const size_t bh = (size_t)(b_ * NHEAD + h);
    const int tid = threadIdx.x, wave = tid >> 5, lane = tid & 31;
    const int hi = lane >> 4, ln = lane & 15;
    const int qw = q0blk + wave * 16;

    const char* gsrc[4];
    unsigned    gstep[4];
    unsigned    lbase[4];
    #pragma unroll
    for (int ii = 0; ii < 4; ++ii) {
        int j = tid + ii * 128;
        if (j < 256) {                        // K^T tile 64x32
            int d = j >> 2, c = j & 3;
            gsrc[ii]  = (const char*)&Kt[(bh * DHEAD + d) * SEQ + c * 8];
            gstep[ii] = 32 * 2;
            lbase[ii] = lds_off(&((uint4*)&Kls[0][0][0])[j]);
        } else {                              // V tile 32x64
            int w = j - 256;
            int r = w >> 3, c = w & 7;
            gsrc[ii]  = (const char*)&Vb[(bh * SEQ + r) * DHEAD + c * 8];
            gstep[ii] = 32 * DHEAD * 2;
            lbase[ii] = lds_off(&((uint4*)&Vls[0][0][0])[w]);
        }
    }
    auto issue = [&](int buf) {               // both sub-buffers are 4096 B
        #pragma unroll
        for (int ii = 0; ii < 4; ++ii) {
            async_ld16_off(lbase[ii] + (buf ? 4096u : 0u), gsrc[ii]);
            gsrc[ii] += gstep[ii];
        }
    };

    v16bf qa[2];
    #pragma unroll
    for (int half = 0; half < 2; ++half)
        #pragma unroll
        for (int e = 0; e < 16; ++e)
            qa[half][e] = Qb[(bh * SEQ + qw + ln) * DHEAD + half * 32 + k_of(e, hi)];

    float mrow[8], lrow[8];
    v8f acc[4];
    #pragma unroll
    for (int r = 0; r < 8; ++r) { mrow[r] = -1e30f; lrow[r] = 0.0f; }
    #pragma unroll
    for (int t = 0; t < 4; ++t)
        #pragma unroll
        for (int r = 0; r < 8; ++r) acc[t][r] = 0.0f;

    const int nsteps = q0blk / 32 + 2;
    issue(0);
    for (int st = 0; st < nsteps; ++st) {
        const int cur = st & 1;
        const int kk0 = st * 32;
        if (st < nsteps - 1) {
            issue(cur ^ 1);
            WAIT_ASYNC(0x4);
        } else {
            WAIT_ASYNC(0x0);
        }
        __syncthreads();

        v8f sc[2];
        #pragma unroll
        for (int ct = 0; ct < 2; ++ct) {
            v8f z;
            #pragma unroll
            for (int r = 0; r < 8; ++r) z[r] = 0.0f;
            v16bf b0, b1;
            #pragma unroll
            for (int e = 0; e < 16; ++e) {
                b0[e] = Kls[cur][k_of(e, hi)][ct * 16 + ln];
                b1[e] = Kls[cur][32 + k_of(e, hi)][ct * 16 + ln];
            }
            z = wmma_bf16(qa[0], b0, z);
            z = wmma_bf16(qa[1], b1, z);
            sc[ct] = z;
        }

        #pragma unroll
        for (int r = 0; r < 8; ++r) {
            int q = qw + r + hi * 8;
            float v0 = sc[0][r] * 0.125f;
            float v1 = sc[1][r] * 0.125f;
            if (kk0 + ln      > q) v0 = -__builtin_inff();
            if (kk0 + 16 + ln > q) v1 = -__builtin_inff();

            float rmax = fmaxf(v0, v1);
            #pragma unroll
            for (int off = 1; off <= 8; off <<= 1) rmax = fmaxf(rmax, __shfl_xor(rmax, off));
            float mnew  = fmaxf(mrow[r], rmax);
            float alpha = __expf(mrow[r] - mnew);
            float p0 = __expf(v0 - mnew);
            float p1 = __expf(v1 - mnew);
            float rsum = p0 + p1;
            #pragma unroll
            for (int off = 1; off <= 8; off <<= 1) rsum += __shfl_xor(rsum, off);
            lrow[r] = lrow[r] * alpha + rsum;
            mrow[r] = mnew;
            #pragma unroll
            for (int t = 0; t < 4; ++t) acc[t][r] *= alpha;
            sc[0][r] = p0;
            sc[1][r] = p1;
        }

        #pragma unroll
        for (int ct = 0; ct < 2; ++ct)
            #pragma unroll
            for (int r = 0; r < 8; ++r)
                Pls[wave][r + hi * 8][ct * 16 + ln] = (__bf16)sc[ct][r];

        v16bf pa;
        #pragma unroll
        for (int e = 0; e < 16; ++e) pa[e] = Pls[wave][ln][k_of(e, hi)];

        #pragma unroll
        for (int t = 0; t < 4; ++t) {
            v16bf vb;
            #pragma unroll
            for (int e = 0; e < 16; ++e) vb[e] = Vls[cur][k_of(e, hi)][t * 16 + ln];
            acc[t] = wmma_bf16(pa, vb, acc[t]);
        }
        __syncthreads();
    }

    #pragma unroll
    for (int t = 0; t < 4; ++t)
        #pragma unroll
        for (int r = 0; r < 8; ++r) {
            int q = qw + r + hi * 8;
            float val = acc[t][r] / lrow[r];
            Ob[((size_t)b_ * SEQ + q) * D_MODEL + h * DHEAD + t * 16 + ln] = (__bf16)val;
        }
}

// ---------------- Phase 3: output projection (f32 result) ----------------

__global__ void __launch_bounds__(128) oproj_kernel(
    const __bf16* __restrict__ Ob, const __bf16* __restrict__ WoT, float* __restrict__ out)
{
    __shared__ alignas(16) __bf16 As[2][64][32];
    __shared__ alignas(16) __bf16 Bs[2][32][64];
    const int row0 = blockIdx.x * 64;
    const int n0   = blockIdx.y * 64;
    const int tid = threadIdx.x, wave = tid >> 5, lane = tid & 31;
    const int hi = lane >> 4, ln = lane & 15;

    const char* gsrc[4];
    unsigned    gstep[4];
    unsigned    lbase[4];
    #pragma unroll
    for (int ii = 0; ii < 4; ++ii) {
        int j = tid + ii * 128;
        if (j < 256) {
            int r = j >> 2, c = j & 3;
            gsrc[ii]  = (const char*)&Ob[(size_t)(row0 + r) * D_MODEL + c * 8];
            gstep[ii] = 32 * 2;
            lbase[ii] = lds_off(&((uint4*)&As[0][0][0])[j]);
        } else {
            int w = j - 256;
            int r = w >> 3, c = w & 7;
            gsrc[ii]  = (const char*)&WoT[(size_t)r * D_MODEL + n0 + c * 8];
            gstep[ii] = 32 * D_MODEL * 2;
            lbase[ii] = lds_off(&((uint4*)&Bs[0][0][0])[w]);
        }
    }
    auto issue = [&](int buf) {
        #pragma unroll
        for (int ii = 0; ii < 4; ++ii) {
            async_ld16_off(lbase[ii] + (buf ? 4096u : 0u), gsrc[ii]);
            gsrc[ii] += gstep[ii];
        }
    };

    v8f acc[4];
    #pragma unroll
    for (int t = 0; t < 4; ++t)
        #pragma unroll
        for (int r = 0; r < 8; ++r) acc[t][r] = 0.0f;

    issue(0);
    for (int kt = 0; kt < D_MODEL / 32; ++kt) {
        const int cur = kt & 1;
        if (kt < D_MODEL / 32 - 1) {
            issue(cur ^ 1);
            WAIT_ASYNC(0x4);
        } else {
            WAIT_ASYNC(0x0);
        }
        __syncthreads();

        v16bf a;
        #pragma unroll
        for (int e = 0; e < 16; ++e) a[e] = As[cur][wave * 16 + ln][k_of(e, hi)];
        #pragma unroll
        for (int t = 0; t < 4; ++t) {
            v16bf b;
            #pragma unroll
            for (int e = 0; e < 16; ++e) b[e] = Bs[cur][k_of(e, hi)][t * 16 + ln];
            acc[t] = wmma_bf16(a, b, acc[t]);
        }
        __syncthreads();
    }

    #pragma unroll
    for (int t = 0; t < 4; ++t)
        #pragma unroll
        for (int r = 0; r < 8; ++r)
            out[(size_t)(row0 + wave * 16 + r + hi * 8) * D_MODEL + n0 + t * 16 + ln] = acc[t][r];
}

// ---------------- launch ----------------

extern "C" void kernel_launch(void* const* d_in, const int* in_sizes, int n_in,
                              void* d_out, int out_size, void* d_ws, size_t ws_size,
                              hipStream_t stream) {
    const float* X   = (const float*)d_in[0];
    const int*   pos = (const int*)  d_in[1];
    const float* Wq  = (const float*)d_in[2];
    const float* Wk  = (const float*)d_in[3];
    const float* Wv  = (const float*)d_in[4];
    const float* Wo  = (const float*)d_in[5];
    float* out = (float*)d_out;

    __bf16* ws = (__bf16*)d_ws;
    const size_t NX  = (size_t)ROWS * D_MODEL;
    const size_t NW  = (size_t)D_MODEL * D_MODEL;
    const size_t NQK = (size_t)BATCH * NHEAD * SEQ * DHEAD;
    __bf16* Xb  = ws;
    __bf16* WqT = Xb  + NX;
    __bf16* WkT = WqT + NW;
    __bf16* WvT = WkT + NW;
    __bf16* WoT = WvT + NW;
    __bf16* Qb  = WoT + NW;
    __bf16* Kt  = Qb  + NQK;
    __bf16* Vb  = Kt  + NQK;
    __bf16* Ob  = Vb  + NQK;

    cvt_bf16_kernel<<<dim3((unsigned)(NX / 8 / 256)), dim3(256), 0, stream>>>(X, Xb);
    transpose_bf16_kernel<<<dim3(32, 32), dim3(32, 32), 0, stream>>>(Wq, WqT);
    transpose_bf16_kernel<<<dim3(32, 32), dim3(32, 32), 0, stream>>>(Wk, WkT);
    transpose_bf16_kernel<<<dim3(32, 32), dim3(32, 32), 0, stream>>>(Wv, WvT);
    transpose_bf16_kernel<<<dim3(32, 32), dim3(32, 32), 0, stream>>>(Wo, WoT);

    qkv_rope_kernel<<<dim3(ROWS / 64, NHEAD), dim3(384), 0, stream>>>(
        Xb, WqT, WkT, WvT, pos, Qb, Kt, Vb);

    attn_kernel<<<dim3(SEQ / 64, NHEAD, BATCH), dim3(128), 0, stream>>>(Qb, Kt, Vb, Ob);

    oproj_kernel<<<dim3(ROWS / 64, D_MODEL / 64), dim3(128), 0, stream>>>(Ob, WoT, out);
}
